// MLA_87540023427078
// MI455X (gfx1250) — compile-verified, matrix operand order checked by reading, not run on hardware
//
#include <hip/hip_runtime.h>
#include <hip/hip_bf16.h>
#include <math.h>

// ---------------- problem constants ----------------
#define B_    4
#define S_    1024
#define DIM_  2048
#define NH_   16
#define QR_   1024
#define KVR_  512
#define NOPE_ 128
#define ROPE_ 64
#define VH_   128
#define QKH_  192
#define DQK_  576            // KVR_ + ROPE_ (concat latent+rope for QK^T)
#define SCALE_ 0.07216878364870323f   // 192^-0.5
#define EPS_  1e-6f

typedef __bf16 bf16_t;
typedef __attribute__((ext_vector_type(16))) __bf16 v16bf;
typedef __attribute__((ext_vector_type(8)))  float  v8f;

// ---------------------------------------------------------------------------
// WMMA fragment gather for NT GEMM (C = A * B^T, A:(M,K) row-major, B:(N,K)
// row-major). Per the CDNA5 ISA 16-bit A layout (16x32, MxK):
//   lane<16 : row = lane,    K = kbase+{0..7} in v[0..3], kbase+{16..23} in v[4..7]
//   lane>=16: row = lane-16, kbase += 8
// The B operand (KxN, N striped across lanes) gathers identically from the
// (N,K) row-major source. Fragment = two contiguous 16B chunks.
// ---------------------------------------------------------------------------
__device__ __forceinline__ v16bf load_frag_nt(const bf16_t* __restrict__ base,
                                              int ld, int rc, int kb) {
  union { uint4 u[2]; v16bf v; } f;
  const uint4* p = (const uint4*)(base + (size_t)rc * ld + kb);
  f.u[0] = p[0];   // K = kb+0 .. kb+7
  f.u[1] = p[2];   // K = kb+16 .. kb+23
  return f.v;
}

// Fragment gather from the LDS-staged B tile (row stride 40 bf16 = 80B, so the
// 16 fragment lanes land on distinct banks). koff in {0,8}.
#define BT_STRIDE 40
__device__ __forceinline__ v16bf frag_lds(const bf16_t* t, int n, int koff) {
  union { uint4 u[2]; v16bf v; } f;
  const uint4* p = (const uint4*)(t + n * BT_STRIDE + koff);
  f.u[0] = p[0];
  f.u[1] = p[2];
  return f.v;
}

#define WMMA_BF16(a, b, c) \
  __builtin_amdgcn_wmma_f32_16x16x32_bf16(false, (a), false, (b), (short)0, (c), false, false)

// ---------------------------------------------------------------------------
// Batched NT GEMM: C[z] = A[z] * B[z%bmod]^T
// Block = 128 threads = 4 waves stacked in M; block tile = 128(M) x 64(N),
// each wave a 32x64 strip (8 accumulators). The shared 64x32 B tile is staged
// per k-step into LDS with GLOBAL_LOAD_ASYNC_TO_LDS_B128 (double buffered,
// s_wait_asynccnt + barrier), A fragments stream directly from global.
// Requires: M % 128 == 0, N % 64 == 0, K % 32 == 0 (true at all call sites).
// ---------------------------------------------------------------------------
__global__ __launch_bounds__(128) void gemm_nt_wmma(
    const bf16_t* __restrict__ A, const bf16_t* __restrict__ Bm,
    void* __restrict__ Cv,
    int M, int N, int K, int lda, int ldb, int ldc,
    long long sA, long long sB, long long sC, int bmod, int out_bf16)
{
  __shared__ __align__(16) bf16_t btile[2][64 * BT_STRIDE];

  const int z = blockIdx.z;
  const bf16_t* Ap = A  + (long long)z * sA;
  const bf16_t* Bp = Bm + (long long)(z % bmod) * sB;

  const int tid   = threadIdx.x;
  const int lane  = tid & 31;
  const int wave  = tid >> 5;
  const int Mbase = blockIdx.y * 128 + wave * 32;
  const int Nbase = blockIdx.x * 64;
  if (Mbase >= M) return;                 // wave-uniform (never taken: M%128==0)

  // Async-stage one 16B chunk of the 64(N) x 32(K) B tile. chunk c in [0,256):
  // row n = c>>2, 8-element k-offset (c&3)*8. 128 threads issue 2 chunks each,
  // unconditionally (no EXEC games around the WMMA region).
  auto stage1 = [&](int buf, int k, int c) {
    const int n  = c >> 2;
    const int kc = (c & 3) << 3;
    unsigned lds_off =
        (unsigned)(uintptr_t)(const void*)&btile[buf][n * BT_STRIDE + kc];
    unsigned long long gaddr =
        (unsigned long long)(uintptr_t)(Bp + (size_t)(Nbase + n) * ldb + (k + kc));
    asm volatile("global_load_async_to_lds_b128 %0, %1, off"
                 :: "v"(lds_off), "v"(gaddr) : "memory");
  };
  auto stage = [&](int buf, int k) {
    stage1(buf, k, tid);
    stage1(buf, k, tid + 128);
  };

  v8f c00 = {}, c01 = {}, c02 = {}, c03 = {};   // rows Mbase..+15
  v8f c10 = {}, c11 = {}, c12 = {}, c13 = {};   // rows Mbase+16..+31

  const int row0 = Mbase + (lane & 15);
  const int row1 = row0 + 16;
  const int koff = (lane >> 4) << 3;
  const int cn   = lane & 15;

  const int nk = K >> 5;
  stage(0, 0);
  for (int ks = 0; ks < nk; ++ks) {
    asm volatile("s_wait_asynccnt 0" ::: "memory");
    __syncthreads();                       // staged tile visible to all waves
    if (ks + 1 < nk) stage((ks + 1) & 1, (ks + 1) << 5);

    const bf16_t* bt = btile[ks & 1];
    const int kb = (ks << 5) + koff;
    v16bf a0 = load_frag_nt(Ap, lda, row0, kb);
    v16bf a1 = load_frag_nt(Ap, lda, row1, kb);
    v16bf b0 = frag_lds(bt,  0 + cn, koff);
    v16bf b1 = frag_lds(bt, 16 + cn, koff);
    v16bf b2 = frag_lds(bt, 32 + cn, koff);
    v16bf b3 = frag_lds(bt, 48 + cn, koff);
    c00 = WMMA_BF16(a0, b0, c00);  c10 = WMMA_BF16(a1, b0, c10);
    c01 = WMMA_BF16(a0, b1, c01);  c11 = WMMA_BF16(a1, b1, c11);
    c02 = WMMA_BF16(a0, b2, c02);  c12 = WMMA_BF16(a1, b2, c12);
    c03 = WMMA_BF16(a0, b3, c03);  c13 = WMMA_BF16(a1, b3, c13);
  }

  // C/D layout: lane<16 -> N=lane, M=Mb+v ; lane>=16 -> N=lane-16, M=Mb+8+v
  const int r0 = Mbase + ((lane >> 4) << 3);
  v8f accs[2][4] = {{c00, c01, c02, c03}, {c10, c11, c12, c13}};
#pragma unroll
  for (int rs = 0; rs < 2; ++rs) {
#pragma unroll
    for (int nt = 0; nt < 4; ++nt) {
      const int col = Nbase + nt * 16 + cn;
#pragma unroll
      for (int v = 0; v < 8; ++v) {
        long long idx = (long long)z * sC + (long long)(r0 + rs * 16 + v) * ldc + col;
        float val = accs[rs][nt][v];
        if (out_bf16) ((bf16_t*)Cv)[idx] = (bf16_t)val;
        else          ((float*) Cv)[idx] = val;
      }
    }
  }
}

// ---------------------------------------------------------------------------
// Elementwise / prep kernels
// ---------------------------------------------------------------------------
__global__ void cvt_bf16_k(const float* __restrict__ in, bf16_t* __restrict__ out,
                           long long n) {
  long long i = (long long)blockIdx.x * blockDim.x + threadIdx.x;
  long long st = (long long)gridDim.x * blockDim.x;
  for (; i < n; i += st) out[i] = (bf16_t)in[i];
}

// wb_kT[h][c][d] = wkv_b[h*256 + d][c]   (h<16, c<512, d<128), bf16
__global__ void wbk_transpose_k(const float* __restrict__ wkvb, bf16_t* __restrict__ out) {
  long long n = (long long)NH_ * 512 * 128;
  long long i = (long long)blockIdx.x * blockDim.x + threadIdx.x;
  long long st = (long long)gridDim.x * blockDim.x;
  for (; i < n; i += st) {
    int d = (int)(i & 127);
    int c = (int)((i >> 7) & 511);
    int h = (int)(i >> 16);
    out[i] = (bf16_t)wkvb[((long long)(h * 256 + d)) * 512 + c];
  }
}

// wb_v[h][d][c] = wkv_b[h*256 + 128 + d][c], bf16
__global__ void wbv_extract_k(const float* __restrict__ wkvb, bf16_t* __restrict__ out) {
  long long n = (long long)NH_ * 128 * 512;
  long long i = (long long)blockIdx.x * blockDim.x + threadIdx.x;
  long long st = (long long)gridDim.x * blockDim.x;
  for (; i < n; i += st) {
    int c = (int)(i & 511);
    int d = (int)((i >> 9) & 127);
    int h = (int)(i >> 16);
    out[i] = (bf16_t)wkvb[((long long)(h * 256 + 128 + d)) * 512 + c];
  }
}

// RMSNorm rows: out[r][c] = bf16(x[r][c] * rsqrt(mean(x^2)+eps) * w[c])
__global__ void rmsnorm_rows_k(const float* __restrict__ X, const float* __restrict__ w,
                               bf16_t* __restrict__ out, int cols) {
  const int r = blockIdx.x;
  const float* x = X + (size_t)r * cols;
  __shared__ float red[256];
  int tid = threadIdx.x;
  float ss = 0.f;
  for (int c = tid; c < cols; c += 256) { float v = x[c]; ss += v * v; }
  red[tid] = ss; __syncthreads();
  for (int o = 128; o > 0; o >>= 1) { if (tid < o) red[tid] += red[tid + o]; __syncthreads(); }
  float inv = rsqrtf(red[0] / (float)cols + EPS_);
  for (int c = tid; c < cols; c += 256) out[(size_t)r * cols + c] = (bf16_t)(x[c] * inv * w[c]);
}

// kv prep: kv_all[r][0:512] = rmsnorm(kvf[r][0:512])*w, kv_all[r][512:576] = rope(...)
__global__ void kv_prep_k(const float* __restrict__ kvf, const float* __restrict__ w,
                          const float* __restrict__ fc, const float* __restrict__ fs,
                          bf16_t* __restrict__ kv_all) {
  const int r = blockIdx.x;
  const int s = r & (S_ - 1);
  const float* x = kvf + (size_t)r * DQK_;
  __shared__ float red[256];
  int tid = threadIdx.x;
  float ss = 0.f;
  for (int c = tid; c < KVR_; c += 256) { float v = x[c]; ss += v * v; }
  red[tid] = ss; __syncthreads();
  for (int o = 128; o > 0; o >>= 1) { if (tid < o) red[tid] += red[tid + o]; __syncthreads(); }
  float inv = rsqrtf(red[0] * (1.0f / KVR_) + EPS_);
  bf16_t* orow = kv_all + (size_t)r * DQK_;
  for (int c = tid; c < KVR_; c += 256) orow[c] = (bf16_t)(x[c] * inv * w[c]);
  if (tid < 32) {
    int i = tid;
    float x0 = x[KVR_ + 2 * i], x1 = x[KVR_ + 2 * i + 1];
    float cs = fc[s * 32 + i], sn = fs[s * 32 + i];
    orow[KVR_ + 2 * i]     = (bf16_t)(x0 * cs - x1 * sn);
    orow[KVR_ + 2 * i + 1] = (bf16_t)(x0 * sn + x1 * cs);
  }
}

// kvT[b][c][t] = kv_all[b*S+t][c]  (c<512) -- V^T for the P*V NT GEMM
__global__ void kvT_k(const bf16_t* __restrict__ kv_all, bf16_t* __restrict__ kvT) {
  long long n = (long long)B_ * KVR_ * S_;
  long long i = (long long)blockIdx.x * blockDim.x + threadIdx.x;
  long long st = (long long)gridDim.x * blockDim.x;
  for (; i < n; i += st) {
    int t = (int)(i & (S_ - 1));
    int c = (int)((i >> 10) & 511);
    int b = (int)(i >> 19);
    kvT[i] = kv_all[((long long)(b * S_ + t)) * DQK_ + c];
  }
}

// qn_b[(b*NH+h)*S+s][d] = q_bf[(b*S+s)*3072 + h*192 + d]  (d<128)
__global__ void q_prep_nope_k(const bf16_t* __restrict__ qb, bf16_t* __restrict__ qn) {
  long long n = (long long)B_ * NH_ * S_ * NOPE_;
  long long i = (long long)blockIdx.x * blockDim.x + threadIdx.x;
  long long st = (long long)gridDim.x * blockDim.x;
  for (; i < n; i += st) {
    int d = (int)(i & 127);
    long long t = i >> 7;
    int s = (int)(t & (S_ - 1)); t >>= 10;
    int h = (int)(t & 15);
    int b = (int)(t >> 4);
    qn[i] = qb[((long long)(b * S_ + s)) * (NH_ * QKH_) + h * QKH_ + d];
  }
}

// q_all[(b*NH+h)*S+s][512+2j(+1)] = rope(q_bf[... h*192+128+2j(+1)])
__global__ void q_prep_rope_k(const bf16_t* __restrict__ qb,
                              const float* __restrict__ fc, const float* __restrict__ fs,
                              bf16_t* __restrict__ q_all) {
  long long n = (long long)B_ * NH_ * S_ * 32;
  long long i = (long long)blockIdx.x * blockDim.x + threadIdx.x;
  long long st = (long long)gridDim.x * blockDim.x;
  for (; i < n; i += st) {
    int j = (int)(i & 31);
    long long t = i >> 5;
    int s = (int)(t & (S_ - 1)); t >>= 10;
    int h = (int)(t & 15);
    int b = (int)(t >> 4);
    long long src = ((long long)(b * S_ + s)) * (NH_ * QKH_) + h * QKH_ + NOPE_ + 2 * j;
    float x0 = (float)qb[src], x1 = (float)qb[src + 1];
    float cs = fc[s * 32 + j], sn = fs[s * 32 + j];
    long long dst = (((long long)(b * NH_ + h)) * S_ + s) * DQK_ + KVR_ + 2 * j;
    q_all[dst]     = (bf16_t)(x0 * cs - x1 * sn);
    q_all[dst + 1] = (bf16_t)(x0 * sn + x1 * cs);
  }
}

// In-place causal softmax over bf16 score rows: P = softmax(scale*scores + causal)
__global__ void softmax_causal_k(bf16_t* __restrict__ SP, float scale) {
  const int sq = blockIdx.x;
  const int h  = blockIdx.y;
  bf16_t* row = SP + ((size_t)h * S_ + sq) * S_;
  __shared__ float red[256];
  int tid = threadIdx.x;
  float m = -3.4e38f;
  for (int t = tid; t < S_; t += 256) {
    float v = (float)row[t] * scale + (t <= sq ? 0.0f : -1.0e9f);
    m = fmaxf(m, v);
  }
  red[tid] = m; __syncthreads();
  for (int o = 128; o > 0; o >>= 1) { if (tid < o) red[tid] = fmaxf(red[tid], red[tid + o]); __syncthreads(); }
  m = red[0]; __syncthreads();
  float s = 0.f;
  for (int t = tid; t < S_; t += 256) {
    float v = (float)row[t] * scale + (t <= sq ? 0.0f : -1.0e9f);
    s += __expf(v - m);
  }
  red[tid] = s; __syncthreads();
  for (int o = 128; o > 0; o >>= 1) { if (tid < o) red[tid] += red[tid + o]; __syncthreads(); }
  float inv = 1.0f / red[0]; __syncthreads();
  for (int t = tid; t < S_; t += 256) {
    float v = (float)row[t] * scale + (t <= sq ? 0.0f : -1.0e9f);
    row[t] = (bf16_t)(__expf(v - m) * inv);
  }
}

// ---------------------------------------------------------------------------
extern "C" void kernel_launch(void* const* d_in, const int* in_sizes, int n_in,
                              void* d_out, int out_size, void* d_ws, size_t ws_size,
                              hipStream_t stream) {
  (void)in_sizes; (void)n_in; (void)out_size; (void)ws_size;
  const float* x        = (const float*)d_in[0];
  /* d_in[1] = mask: causal mask applied analytically */
  const float* fcos     = (const float*)d_in[2];
  const float* fsin     = (const float*)d_in[3];
  const float* wq_a     = (const float*)d_in[4];
  const float* q_norm_w = (const float*)d_in[5];
  const float* wq_b     = (const float*)d_in[6];
  const float* wkv_a    = (const float*)d_in[7];
  const float* kv_norm_w= (const float*)d_in[8];
  const float* wkv_b    = (const float*)d_in[9];
  const float* wo       = (const float*)d_in[10];
  float* out = (float*)d_out;

  char* ws = (char*)d_ws;
  size_t off = 0;
  auto carve = [&](size_t bytes) -> void* {
    void* p = ws + off; off += (bytes + 255) & ~(size_t)255; return p;
  };

  const long long nBS = (long long)B_ * S_;                 // 4096
  bf16_t* xb     = (bf16_t*)carve(nBS * DIM_ * 2);          // x, bf16
  bf16_t* wqa_b  = (bf16_t*)carve((long long)QR_ * DIM_ * 2);
  bf16_t* wqb_b  = (bf16_t*)carve((long long)NH_ * QKH_ * QR_ * 2);
  bf16_t* wkva_b = (bf16_t*)carve((long long)DQK_ * DIM_ * 2);
  bf16_t* wbkT   = (bf16_t*)carve((long long)NH_ * KVR_ * NOPE_ * 2);
  bf16_t* wbv    = (bf16_t*)carve((long long)NH_ * VH_ * KVR_ * 2);
  bf16_t* wo_b   = (bf16_t*)carve((long long)DIM_ * DIM_ * 2);
  float*  xq     = (float*) carve(nBS * QR_ * 4);           // x@wq_a^T
  bf16_t* xqn    = (bf16_t*)carve(nBS * QR_ * 2);           // rmsnorm(xq)
  bf16_t* q_bf   = (bf16_t*)carve(nBS * NH_ * QKH_ * 2);    // q
  float*  kvf    = (float*) carve(nBS * DQK_ * 4);          // x@wkv_a^T
  bf16_t* kv_all = (bf16_t*)carve(nBS * DQK_ * 2);          // [kv_norm | k_pe] (B,S,576)
  bf16_t* kvT    = (bf16_t*)carve((long long)B_ * KVR_ * S_ * 2); // (B,512,S)
  bf16_t* qn_b   = (bf16_t*)carve((long long)B_ * NH_ * S_ * NOPE_ * 2); // (B,NH,S,128)
  bf16_t* q_all  = (bf16_t*)carve((long long)B_ * NH_ * S_ * DQK_ * 2);  // (B,NH,S,576)
  bf16_t* SP     = (bf16_t*)carve((long long)NH_ * S_ * S_ * 2);   // scores/probs, per-b reuse
  bf16_t* o_head = (bf16_t*)carve((long long)NH_ * B_ * S_ * KVR_ * 2);  // (NH,B,S,512)
  bf16_t* o2     = (bf16_t*)carve(nBS * DIM_ * 2);          // (B*S, NH*VH)

  auto gemm = [&](const bf16_t* A, const bf16_t* Bm, void* C, int M, int N, int K,
                  int lda, int ldb, int ldc, long long sA, long long sB, long long sC,
                  int Z, int bmod, int obf) {
    dim3 g((unsigned)(N / 64), (unsigned)(M / 128), (unsigned)Z);
    gemm_nt_wmma<<<g, 128, 0, stream>>>(A, Bm, C, M, N, K, lda, ldb, ldc, sA, sB, sC, bmod, obf);
  };
  auto cvt = [&](const float* in, bf16_t* o, long long n) {
    cvt_bf16_k<<<2048, 256, 0, stream>>>(in, o, n);
  };

  // 0) precision conversion of activations + weights
  cvt(x,     xb,     nBS * DIM_);
  cvt(wq_a,  wqa_b,  (long long)QR_ * DIM_);
  cvt(wq_b,  wqb_b,  (long long)NH_ * QKH_ * QR_);
  cvt(wkv_a, wkva_b, (long long)DQK_ * DIM_);
  cvt(wo,    wo_b,   (long long)DIM_ * DIM_);
  wbk_transpose_k<<<2048, 256, 0, stream>>>(wkv_b, wbkT);
  wbv_extract_k  <<<2048, 256, 0, stream>>>(wkv_b, wbv);

  // 1) xq = x @ wq_a^T   (f32 out for rmsnorm precision)
  gemm(xb, wqa_b, xq, (int)nBS, QR_, DIM_, DIM_, DIM_, QR_, 0, 0, 0, 1, 1, 0);
  // 2) rmsnorm -> bf16
  rmsnorm_rows_k<<<(unsigned)nBS, 256, 0, stream>>>(xq, q_norm_w, xqn, QR_);
  // 3) q = xqn @ wq_b^T  (bf16 out)
  gemm(xqn, wqb_b, q_bf, (int)nBS, NH_ * QKH_, QR_, QR_, QR_, NH_ * QKH_, 0, 0, 0, 1, 1, 1);
  // 4) kv_full = x @ wkv_a^T (f32 out)
  gemm(xb, wkva_b, kvf, (int)nBS, DQK_, DIM_, DIM_, DIM_, DQK_, 0, 0, 0, 1, 1, 0);
  // 5) kv prep (rmsnorm + rope) and V^T
  kv_prep_k<<<(unsigned)nBS, 256, 0, stream>>>(kvf, kv_norm_w, fcos, fsin, kv_all);
  kvT_k<<<2048, 256, 0, stream>>>(kv_all, kvT);
  // 6) q prep: extract q_nope, rope q_pe into q_all[...,512:576]
  q_prep_nope_k<<<2048, 256, 0, stream>>>(q_bf, qn_b);
  q_prep_rope_k<<<2048, 256, 0, stream>>>(q_bf, fcos, fsin, q_all);
  // 7) q_lat = q_nope @ wb_k  -> q_all[...,0:512]; z = b*NH+h, weights repeat mod NH
  gemm(qn_b, wbkT, q_all, S_, KVR_, NOPE_, NOPE_, NOPE_, DQK_,
       (long long)S_ * NOPE_, (long long)KVR_ * NOPE_, (long long)S_ * DQK_,
       B_ * NH_, NH_, 1);

  // 8) attention per batch b (z = head)
  for (int b = 0; b < B_; ++b) {
    // scores = Q'(S,576) @ K'(S,576)^T -> SP (bf16)
    gemm(q_all + (long long)b * NH_ * S_ * DQK_, kv_all + (long long)b * S_ * DQK_, SP,
         S_, S_, DQK_, DQK_, DQK_, S_,
         (long long)S_ * DQK_, 0, (long long)S_ * S_, NH_, 1, 1);
    // causal softmax in place
    softmax_causal_k<<<dim3(S_, NH_), 256, 0, stream>>>(SP, SCALE_);
    // O = P(S,S) @ V(S,512)  via NT with kvT(512,S)
    gemm(SP, kvT + (long long)b * KVR_ * S_, o_head + (long long)b * S_ * KVR_,
         S_, KVR_, S_, S_, S_, KVR_,
         (long long)S_ * S_, 0, (long long)B_ * S_ * KVR_, NH_, 1, 1);
  }

  // 9) o2[:, h*128:(h+1)*128] = o_head[h] @ wb_v[h]^T   (z = head)
  gemm(o_head, wbv, o2, (int)nBS, VH_, KVR_, KVR_, KVR_, DIM_,
       (long long)B_ * S_ * KVR_, (long long)VH_ * KVR_, (long long)VH_,
       NH_, NH_, 1);
  // 10) out = o2 @ wo^T (f32 out -> d_out)
  gemm(o2, wo_b, out, (int)nBS, DIM_, DIM_, DIM_, DIM_, DIM_, 0, 0, 0, 1, 1, 0);
}
